// MultiHeadAttention_6081673691577
// MI455X (gfx1250) — matrix operand order repaired and checked by reading, not compiled, not run on hardware
//
#include <hip/hip_runtime.h>

// ---------------------------------------------------------------------------
// MultiHeadAttention for MI455X (gfx1250, wave32, WMMA).
// All GEMMs use v_wmma_f32_16x16x32_bf16 (f32 accumulate), software-pipelined
// with ping-pong fragment buffers in a ROLLED loop (backedge forces the
// allocator to keep a full k-step of fragments in flight), unrolled x2 so
// most phase-to-phase dataflow is direct SSA (halves the phi-copy overhead).
// Sizes fixed by the reference: B=2 N=2048 D=1024 H=16 E=64 O=1024.
// ---------------------------------------------------------------------------

#define B_   2
#define N_   2048
#define D_   1024
#define H_   16
#define E_   64
#define O_   1024
#define HE_  (H_ * E_)      // 1024
#define ROWS (B_ * N_)      // 4096

typedef __attribute__((ext_vector_type(16))) __bf16 v16bf;
typedef __attribute__((ext_vector_type(8)))  __bf16 v8bf;
typedef __attribute__((ext_vector_type(8)))  float  v8f;
typedef __attribute__((ext_vector_type(4)))  float  f32x4;

// 16 contiguous bf16 (32B) -> v16bf  (B-matrix fragment: lane n=l&15, K=16*(l>>4)+j)
__device__ __forceinline__ v16bf load_bf16x16_contig(const __bf16* p) {
    union { f32x4 q[2]; v16bf v; } u;
    u.q[0] = *(const f32x4*)(p);
    u.q[1] = *(const f32x4*)(p + 8);
    return u.v;
}
// two 8-bf16 chunks (A-matrix fragment halves) -> v16bf
__device__ __forceinline__ v16bf load_bf16x8x2(const __bf16* p0, const __bf16* p1) {
    union { f32x4 q[2]; v16bf v; } u;
    u.q[0] = *(const f32x4*)(p0);
    u.q[1] = *(const f32x4*)(p1);
    return u.v;
}

__device__ __forceinline__ v8f wmma_bf16(v16bf a, v16bf b, v8f c) {
    return __builtin_amdgcn_wmma_f32_16x16x32_bf16(false, a, false, b,
                                                   (short)0, c, false, false);
}

// One wave computes C[16 x 64] = A[16 x K] * W[K x 64], W given transposed (WT[64 x K]).
// A row-major bf16 (lda = K elements), WT row-major bf16 (ldw = K elements).
// Rolled ping-pong double buffering (K multiple of 128 here: 1024):
// each phase consumes one buffer and refills it for k-chunk s+64 while the
// other buffer's WMMAs and the loop backedge cover the latency.
__device__ __forceinline__ void wave_gemm_16x64(const __bf16* __restrict__ A, int lda,
                                                const __bf16* __restrict__ WT, int ldw,
                                                const int K, int lm, int lh, v8f acc[4]) {
    const __bf16* arow = A  + (size_t)lm * lda + 16 * lh;
    const __bf16* wrow = WT + (size_t)lm * ldw + 16 * lh;

    v16bf af0 = load_bf16x8x2(arow, arow + 8);
    v16bf af1 = load_bf16x8x2(arow + 32, arow + 40);
    v16bf wf0[4], wf1[4];
#pragma unroll
    for (int e = 0; e < 4; ++e) {
        wf0[e] = load_bf16x16_contig(wrow + (size_t)(16 * e) * ldw);
        wf1[e] = load_bf16x16_contig(wrow + (size_t)(16 * e) * ldw + 32);
    }

#pragma unroll 2
    for (int s = 0; s < K; s += 64) {
        // clamped prefetch: final refills re-read s=0/32 and are discarded (no OOB)
        const int s2 = (s + 64 < K) ? s + 64 : 0;

        // phase 0: consume buf0 (k-chunk s), refill buf0 with k-chunk s2
#pragma unroll
        for (int e = 0; e < 4; ++e)
            acc[e] = wmma_bf16(af0, wf0[e], acc[e]);
        af0 = load_bf16x8x2(arow + s2, arow + s2 + 8);
#pragma unroll
        for (int e = 0; e < 4; ++e)
            wf0[e] = load_bf16x16_contig(wrow + (size_t)(16 * e) * ldw + s2);

        // phase 1: consume buf1 (k-chunk s+32), refill buf1 with k-chunk s2+32
#pragma unroll
        for (int e = 0; e < 4; ++e)
            acc[e] = wmma_bf16(af1, wf1[e], acc[e]);
        af1 = load_bf16x8x2(arow + s2 + 32, arow + s2 + 40);
#pragma unroll
        for (int e = 0; e < 4; ++e)
            wf1[e] = load_bf16x16_contig(wrow + (size_t)(16 * e) * ldw + s2 + 32);
    }
}

// ------------------------------- prep kernels ------------------------------

__global__ void cvt_bf16_kernel(const float* __restrict__ in, __bf16* __restrict__ out, int n) {
    int i = blockIdx.x * blockDim.x + threadIdx.x;
    if (i < n) out[i] = (__bf16)in[i];
}

// out[z][c][r] = in[z][r][c]   (batched transpose + f32->bf16)
__global__ void transpose_bf16_kernel(const float* __restrict__ in, __bf16* __restrict__ out,
                                      int rows, int cols) {
    int z = blockIdx.z;
    int i = blockIdx.x * blockDim.x + threadIdx.x;
    int total = rows * cols;
    if (i >= total) return;
    int r = i / cols, c = i - r * cols;
    out[(size_t)z * total + (size_t)c * rows + r] = (__bf16)in[(size_t)z * total + (size_t)r * cols + c];
}

// ---------------------------- QKV projections ------------------------------

// out[b][h][n][e] = alpha * sum_d X[b*N+n][d] * W[h][d][e]   (WT = W[h] transposed [E x D])
__global__ void __launch_bounds__(128)
qk_proj_kernel(const __bf16* __restrict__ Xb, const __bf16* __restrict__ WT,
               __bf16* __restrict__ out, float alpha) {
    const int wave = threadIdx.x >> 5, lane = threadIdx.x & 31;
    const int lm = lane & 15, lh = lane >> 4;
    const int h  = blockIdx.y;
    const int rb = blockIdx.x * 64 + wave * 16;

    v8f zero8 = {0.f, 0.f, 0.f, 0.f, 0.f, 0.f, 0.f, 0.f};
    v8f acc[4] = {zero8, zero8, zero8, zero8};
    wave_gemm_16x64(Xb + (size_t)rb * D_, D_, WT + (size_t)h * E_ * D_, D_, D_, lm, lh, acc);

    const int b = rb >> 11;              // N_ == 2048
#pragma unroll
    for (int e = 0; e < 4; ++e) {
#pragma unroll
        for (int r = 0; r < 8; ++r) {
            int n = (rb & (N_ - 1)) + r + 8 * lh;
            out[(((size_t)(b * H_ + h) * N_ + n) * E_) + 16 * e + lm] = (__bf16)(acc[e][r] * alpha);
        }
    }
}

// vT[b][h][e][m] = sum_d X[b*N+m][d] * Wv[h][d][e]  (transposed store, packed v8bf)
__global__ void __launch_bounds__(128)
v_proj_kernel(const __bf16* __restrict__ Xb, const __bf16* __restrict__ WT,
              __bf16* __restrict__ vT) {
    const int wave = threadIdx.x >> 5, lane = threadIdx.x & 31;
    const int lm = lane & 15, lh = lane >> 4;
    const int h  = blockIdx.y;
    const int rb = blockIdx.x * 64 + wave * 16;

    v8f zero8 = {0.f, 0.f, 0.f, 0.f, 0.f, 0.f, 0.f, 0.f};
    v8f acc[4] = {zero8, zero8, zero8, zero8};
    wave_gemm_16x64(Xb + (size_t)rb * D_, D_, WT + (size_t)h * E_ * D_, D_, D_, lm, lh, acc);

    const int b  = rb >> 11;
    const int n0 = (rb & (N_ - 1)) + 8 * lh;   // 8 consecutive m per lane
#pragma unroll
    for (int e = 0; e < 4; ++e) {
        v8bf o = __builtin_convertvector(acc[e], v8bf);
        *(v8bf*)(vT + ((size_t)(b * H_ + h) * E_ + 16 * e + lm) * N_ + n0) = o;
    }
}

// ------------------------------ flash attention ----------------------------
// One wave owns 16 q-rows. GEMM1 computes S^T = K_tile(32x64) * q^T(64x16);
// its C layout equals the A-fragment layout of P for GEMM2 O += P(16x32)*V(32x64),
// so exp()+bf16 pack needs no cross-lane movement.
// One step processes 32 keys at M0 using ka_cur and refills ka_nxt for MN;
// the refill + V loads fly under the softmax VALU/transcendental work.
__device__ __forceinline__ void attn_step(const __bf16* __restrict__ kp,
                                          const __bf16* __restrict__ vp,
                                          int M0, int MN,
                                          v16bf (&ka_cur)[2][2], v16bf (&ka_nxt)[2][2],
                                          const v16bf (&qb)[2], v8f (&acc)[4],
                                          float& run_max, float& run_sum, int lh) {
    v8f zero8 = {0.f, 0.f, 0.f, 0.f, 0.f, 0.f, 0.f, 0.f};
    // S^T[32 k x 16 q] in two C fragments
    v8f st[2] = {zero8, zero8};
#pragma unroll
    for (int mi = 0; mi < 2; ++mi)
#pragma unroll
        for (int s = 0; s < 2; ++s)
            st[mi] = wmma_bf16(ka_cur[mi][s], qb[s], st[mi]);

    // refill the other ping-pong buffer with the next key tile
#pragma unroll
    for (int mi = 0; mi < 2; ++mi)
#pragma unroll
        for (int s = 0; s < 2; ++s) {
            const __bf16* kr = kp + (size_t)(MN + 16 * mi) * E_ + 32 * s;
            ka_nxt[mi][s] = load_bf16x8x2(kr, kr + 8);
        }
    // this tile's V fragments (consumed at the end of the step)
    v16bf vb[4];
#pragma unroll
    for (int e = 0; e < 4; ++e)
        vb[e] = load_bf16x8x2(vp + (size_t)(16 * e) * N_ + M0, vp + (size_t)(16 * e) * N_ + M0 + 16);

    // online softmax over k (rows of S^T): in-lane over 8 VGPRs x 2 frags + halves
    float tmax = run_max;
#pragma unroll
    for (int mi = 0; mi < 2; ++mi)
#pragma unroll
        for (int r = 0; r < 8; ++r) tmax = fmaxf(tmax, st[mi][r]);
    tmax = fmaxf(tmax, __shfl_xor(tmax, 16, 32));
    float corr = __expf(run_max - tmax);

    v16bf pa;                        // P as a ready-made A-fragment
    float psum = 0.f;
#pragma unroll
    for (int r = 0; r < 8; ++r) {
        float e0 = __expf(st[0][r] - tmax);
        float e1 = __expf(st[1][r] - tmax);
        psum += e0 + e1;
        pa[r]     = (__bf16)e0;
        pa[r + 8] = (__bf16)e1;
    }
    psum += __shfl_xor(psum, 16, 32);
    run_sum = run_sum * corr + psum;
    run_max = tmax;

    // rescale O: factor for O-row (r + 8*lh) lives on lane (r + 8*lh)
#pragma unroll
    for (int r = 0; r < 8; ++r) {
        float cr = __shfl(corr, r + 8 * lh, 32);
#pragma unroll
        for (int e = 0; e < 4; ++e) acc[e][r] *= cr;
    }
    // O += P * V
#pragma unroll
    for (int e = 0; e < 4; ++e)
        acc[e] = wmma_bf16(pa, vb[e], acc[e]);
}

__global__ void __launch_bounds__(128)
attn_kernel(const __bf16* __restrict__ q, const __bf16* __restrict__ k,
            const __bf16* __restrict__ vT, __bf16* __restrict__ out) {
    const int wave = threadIdx.x >> 5, lane = threadIdx.x & 31;
    const int lm = lane & 15, lh = lane >> 4;
    const int h = blockIdx.y, b = blockIdx.z;
    const int nq0 = (blockIdx.x * 4 + wave) * 16;

    const __bf16* qp = q  + ((size_t)(b * H_ + h) * N_ + nq0) * E_;   // [16 x 64]
    const __bf16* kp = k  +  (size_t)(b * H_ + h) * N_ * E_           // [2048 x 64]
                          + (size_t)lm * E_ + 16 * lh;                //  per-lane base
    const __bf16* vp = vT +  (size_t)(b * H_ + h) * E_ * N_           // [64 x 2048]
                          + (size_t)lm * N_ + 8 * lh;                 //  per-lane base

    // q as B-fragments (reused across the whole key loop)
    v16bf qb[2];
#pragma unroll
    for (int s = 0; s < 2; ++s)
        qb[s] = load_bf16x16_contig(qp + lm * E_ + 32 * s + 16 * lh);

    v8f zero8 = {0.f, 0.f, 0.f, 0.f, 0.f, 0.f, 0.f, 0.f};
    v8f acc[4] = {zero8, zero8, zero8, zero8};          // O[16 x 64]
    float run_max = -1e30f, run_sum = 0.f;              // per q-col (= lane&15)

    // preload key tile m0 = 0 into ping buffer
    v16bf ka0[2][2], ka1[2][2];
#pragma unroll
    for (int mi = 0; mi < 2; ++mi)
#pragma unroll
        for (int s = 0; s < 2; ++s) {
            const __bf16* kr = kp + (size_t)(16 * mi) * E_ + 32 * s;
            ka0[mi][s] = load_bf16x8x2(kr, kr + 8);
        }

#pragma unroll 2
    for (int m0 = 0; m0 < N_; m0 += 64) {
        // half-step A: keys [m0, m0+32), prefetch [m0+32, m0+64) into ka1
        attn_step(kp, vp, m0, m0 + 32, ka0, ka1, qb, acc, run_max, run_sum, lh);
        // half-step B: keys [m0+32, m0+64), prefetch next iter (clamped) into ka0
        int mn = (m0 + 64 < N_) ? m0 + 64 : 0;
        attn_step(kp, vp, m0 + 32, mn, ka1, ka0, qb, acc, run_max, run_sum, lh);
    }

    const float inv = 1.f / run_sum;
#pragma unroll
    for (int r = 0; r < 8; ++r) {
        float ir = __shfl(inv, r + 8 * lh, 32);
        int n = nq0 + r + 8 * lh;
#pragma unroll
        for (int e = 0; e < 4; ++e) {
            // interleaved feature order: feat = e_global*H + h   (E-major, per reference)
            out[(size_t)(b * N_ + n) * HE_ + (16 * e + lm) * H_ + h] = (__bf16)(acc[e][r] * ir);
        }
    }
}

// ---------------------------- output projection ----------------------------

__global__ void __launch_bounds__(128)
out_proj_kernel(const __bf16* __restrict__ Ab, const __bf16* __restrict__ WpT,
                const float* __restrict__ bp, float* __restrict__ out) {
    const int wave = threadIdx.x >> 5, lane = threadIdx.x & 31;
    const int lm = lane & 15, lh = lane >> 4;
    const int rb = blockIdx.x * 64 + wave * 16;
    const int cb = blockIdx.y * 64;

    v8f zero8 = {0.f, 0.f, 0.f, 0.f, 0.f, 0.f, 0.f, 0.f};
    v8f acc[4] = {zero8, zero8, zero8, zero8};
    wave_gemm_16x64(Ab + (size_t)rb * HE_, HE_, WpT + (size_t)cb * HE_, HE_, HE_, lm, lh, acc);

#pragma unroll
    for (int e = 0; e < 4; ++e) {
        float bias = bp[cb + 16 * e + lm];
#pragma unroll
        for (int r = 0; r < 8; ++r) {
            out[(size_t)(rb + r + 8 * lh) * O_ + cb + 16 * e + lm] = acc[e][r] + bias;
        }
    }
}

// --------------------------------- launch ----------------------------------

extern "C" void kernel_launch(void* const* d_in, const int* in_sizes, int n_in,
                              void* d_out, int out_size, void* d_ws, size_t ws_size,
                              hipStream_t stream) {
    (void)in_sizes; (void)n_in; (void)out_size; (void)ws_size;
    const float* query = (const float*)d_in[0];
    const float* Wq    = (const float*)d_in[1];
    const float* Wk    = (const float*)d_in[2];
    const float* Wv    = (const float*)d_in[3];
    const float* Wp    = (const float*)d_in[4];
    const float* bp    = (const float*)d_in[5];

    __bf16* base = (__bf16*)d_ws;
    size_t o = 0;
    __bf16* Xb   = base + o; o += (size_t)ROWS * D_;      // 4096x1024
    __bf16* WqT  = base + o; o += (size_t)H_ * E_ * D_;   // 16x[64x1024]
    __bf16* WkT  = base + o; o += (size_t)H_ * E_ * D_;
    __bf16* WvT  = base + o; o += (size_t)H_ * E_ * D_;
    __bf16* WpT  = base + o; o += (size_t)O_ * HE_;       // [1024x1024]
    __bf16* qbuf = base + o; o += (size_t)B_ * H_ * N_ * E_;
    __bf16* kbuf = base + o; o += (size_t)B_ * H_ * N_ * E_;
    __bf16* vTb  = base + o; o += (size_t)B_ * H_ * E_ * N_;
    __bf16* ab   = base + o; o += (size_t)ROWS * HE_;     // attention out, interleaved

    const int nX = ROWS * D_;
    cvt_bf16_kernel<<<(nX + 255) / 256, 256, 0, stream>>>(query, Xb, nX);
    transpose_bf16_kernel<<<dim3((D_ * E_ + 255) / 256, 1, H_), 256, 0, stream>>>(Wq, WqT, D_, E_);
    transpose_bf16_kernel<<<dim3((D_ * E_ + 255) / 256, 1, H_), 256, 0, stream>>>(Wk, WkT, D_, E_);
    transpose_bf16_kernel<<<dim3((D_ * E_ + 255) / 256, 1, H_), 256, 0, stream>>>(Wv, WvT, D_, E_);
    transpose_bf16_kernel<<<dim3((HE_ * O_ + 255) / 256, 1, 1), 256, 0, stream>>>(Wp, WpT, HE_, O_);

    // scale 1/sqrt(E) folded into the Q projection
    qk_proj_kernel<<<dim3(ROWS / 64, H_), 128, 0, stream>>>(Xb, WqT, qbuf, 0.125f);
    qk_proj_kernel<<<dim3(ROWS / 64, H_), 128, 0, stream>>>(Xb, WkT, kbuf, 1.0f);
    v_proj_kernel <<<dim3(ROWS / 64, H_), 128, 0, stream>>>(Xb, WvT, vTb);

    attn_kernel<<<dim3(N_ / 64, H_, B_), 128, 0, stream>>>(qbuf, kbuf, vTb, ab);

    out_proj_kernel<<<dim3(ROWS / 64, O_ / 64), 128, 0, stream>>>(ab, WpT, bp, (float*)d_out);
}
